// INRF_27814208209269
// MI455X (gfx1250) — matrix operand order repaired and checked by reading, not compiled, not run on hardware
//
#include <hip/hip_runtime.h>

// ---- CDNA5 vector types for WMMA ----
typedef __attribute__((ext_vector_type(16))) __bf16 v16bf;
typedef __attribute__((ext_vector_type(8)))  __bf16 v8bf;
typedef __attribute__((ext_vector_type(8)))  float  v8f;

constexpr int NB = 8;       // batch
constexpr int NH = 256;     // height
constexpr int NW = 256;     // width
constexpr int NC = 16;      // channels
constexpr int KSZ = 21;     // kernel size
constexpr int NTAPS = KSZ * KSZ;        // 441
constexpr int NPAIRS = (NTAPS + 1) / 2; // 221 (last pair half-zero)
constexpr int PR = 28;      // patch rows  (8 out rows + 20 halo)
constexpr int PCW = 36;     // patch cols  (16 out cols + 20 halo)
constexpr int PSITES = PR * PCW;        // 1008 pixel sites per patch

__device__ __forceinline__ float d2of(int i) {
  float d = (float)(i - 10) * (1.0f / 32.0f);   // x = i/32, mean = 10/32
  return d * d;
}

// ---------------- Phase 1a: per-(cin,cout) inverse L2 norm ----------------
// Normalized kernel = exp(-0.5 g^2 r2) / sqrt(sum_t exp(-g^2 r2_t))  (g^2/2pi cancels)
__global__ void k_norms(const float* __restrict__ gm, const float* __restrict__ gw,
                        float* __restrict__ normInv) {
  int wsel = blockIdx.x;          // 0 = gamma_m, 1 = gamma_w
  int tid = threadIdx.x;          // cin*16 + cout
  float g = (wsel == 0 ? gm : gw)[tid];
  float g2 = g * g;
  float sum = 0.0f;
  for (int t = 0; t < NTAPS; ++t) {
    int kh = t / KSZ, kw = t % KSZ;
    float e = __expf(-0.5f * g2 * (d2of(kh) + d2of(kw)));
    sum += e * e;
  }
  normInv[wsel * 256 + tid] = rsqrtf(sum);
}

// ---------------- Phase 1b: weights in WMMA B-fragment order ----------------
// frag[wsel][p][lane][j]: lane = n + 16*half stores (K = 16*half + j, N = lane%16)
//   -> tap = 2p + half, cin = j, cout = lane%16. tap 441 => 0.
__global__ void k_frags(const float* __restrict__ gm, const float* __restrict__ gw,
                        const float* __restrict__ normInv, __bf16* __restrict__ frag) {
  int idx = blockIdx.x * blockDim.x + threadIdx.x;
  if (idx >= 2 * NPAIRS * 512) return;
  int wsel = idx / (NPAIRS * 512);
  int rem  = idx % (NPAIRS * 512);
  int p    = rem / 512;
  int e    = rem % 512;
  int lane = e / 16;
  int j    = e % 16;        // = cin
  int n    = lane & 15;     // cout
  int half = lane >> 4;
  int t    = 2 * p + half;  // tap index
  float val = 0.0f;
  if (t < NTAPS) {
    float g = (wsel == 0 ? gm : gw)[j * 16 + n];
    int kh = t / KSZ, kw = t % KSZ;
    val = __expf(-0.5f * g * g * (d2of(kh) + d2of(kw))) * normInv[wsel * 256 + j * 16 + n];
  }
  frag[idx] = (__bf16)val;
}

// ---------------- Phase 2a: channel sums, row sums, bf16 convert ----------------
__global__ void k_stats1(const float* __restrict__ x, __bf16* __restrict__ xbf,
                         float* __restrict__ s, float* __restrict__ R) {
  int y = blockIdx.x, b = blockIdx.y, xc = threadIdx.x;
  size_t base = (((size_t)b * NH + y) * NW + xc) * NC;
  const float* px = x + base;
  __bf16* pb = xbf + base;
  float sum = 0.0f;
#pragma unroll
  for (int c = 0; c < NC; ++c) { float v = px[c]; sum += v; pb[c] = (__bf16)v; }
  s[((size_t)b * NH + y) * NW + xc] = sum;
  __shared__ float red[256];
  red[xc] = sum;
  __syncthreads();
  for (int st = 128; st > 0; st >>= 1) {
    if (xc < st) red[xc] += red[xc + st];
    __syncthreads();
  }
  if (xc == 0) R[b * NH + y] = red[0];
}

// ---------------- Phase 2b: column sums + total ----------------
__global__ void k_stats2(const float* __restrict__ s, float* __restrict__ Cc,
                         float* __restrict__ T) {
  int b = blockIdx.x, xc = threadIdx.x;
  float sum = 0.0f;
  for (int y = 0; y < NH; ++y) sum += s[((size_t)b * NH + y) * NW + xc];
  Cc[b * NW + xc] = sum;
  __shared__ float red[256];
  red[xc] = sum;
  __syncthreads();
  for (int st = 128; st > 0; st >>= 1) {
    if (xc < st) red[xc] += red[xc + st];
    __syncthreads();
  }
  if (xc == 0) T[b] = red[0];
}

// ---------------- Phase 2c: difference term (wrap-window closed form) ----------------
// win(y,x) = T - R[(y+128)%256] - C[(x+128)%256] + s[(y+128)%256,(x+128)%256]
// diff = 2*x - win (broadcast over channels)
__global__ void k_diff(const float* __restrict__ x, const float* __restrict__ s,
                       const float* __restrict__ R, const float* __restrict__ Cc,
                       const float* __restrict__ T, __bf16* __restrict__ dbf) {
  int y = blockIdx.x, b = blockIdx.y, xc = threadIdx.x;
  int y2 = (y + 128) & 255, x2 = (xc + 128) & 255;
  float win = T[b] - R[b * NH + y2] - Cc[b * NW + x2] + s[((size_t)b * NH + y2) * NW + x2];
  size_t base = (((size_t)b * NH + y) * NW + xc) * NC;
  const float* px = x + base;
  __bf16* pd = dbf + base;
#pragma unroll
  for (int c = 0; c < NC; ++c) pd[c] = (__bf16)(2.0f * px[c] - win);
}

// ---------------- Phase 3: fused implicit-GEMM convs via v_wmma_f32_16x16x32_bf16 ----
// out = convM(x) - convW(diff), both 21x21 16->16ch, computed in one pass.
// Block: 256 threads = 8 waves; tile 16 x-pixels wide, 8 y-rows tall.
// Wave w computes output row y0+w: two D[16 px x 16 cout] accumulators,
// 221 tap-pair iterations, 2 WMMAs each (one per conv). Single write pass.
__global__ __launch_bounds__(256) void k_conv_fused(
    const __bf16* __restrict__ srcX, const __bf16* __restrict__ srcD,
    const __bf16* __restrict__ fragM, const __bf16* __restrict__ fragW,
    float* __restrict__ out) {
  // two halo patches: [0] = x, [1] = diff ; 2 * 32,256 B = 63 KB LDS
  __shared__ __align__(16) __bf16 patch[2 * PSITES * NC];
  int tid = threadIdx.x;
  int x0 = blockIdx.x * 16;
  int y0 = blockIdx.y * 8;
  int b  = blockIdx.z;

  // Cooperative stage of both (28 x 36 x 16ch) bf16 halo patches (zero "SAME" pad).
  for (int i = tid; i < 2 * PSITES; i += 256) {
    int which = i >= PSITES;
    int site  = which ? i - PSITES : i;
    int r = site / PCW, cc = site % PCW;
    int gy = y0 - 10 + r, gx = x0 - 10 + cc;
    uint4* lp = reinterpret_cast<uint4*>(&patch[i * NC]);
    if ((unsigned)gy < (unsigned)NH && (unsigned)gx < (unsigned)NW) {
      const __bf16* g = (which ? srcD : srcX) + (((size_t)b * NH + gy) * NW + gx) * NC;
      const uint4* gp = reinterpret_cast<const uint4*>(g);
      lp[0] = gp[0];
      lp[1] = gp[1];
    } else {
      uint4 z; z.x = z.y = z.z = z.w = 0u;
      lp[0] = z;
      lp[1] = z;
    }
  }
  __syncthreads();

  int wv   = tid >> 5;    // wave id = output-row offset (0..7)
  int lane = tid & 31;
  int m    = lane & 15;   // A row (pixel) index; also D col (cout) index
  int half = lane >> 4;

  const __bf16* patchD = &patch[PSITES * NC];

  v8f accM = {0.f, 0.f, 0.f, 0.f, 0.f, 0.f, 0.f, 0.f};
  v8f accW = accM;

  for (int p = 0; p < NPAIRS - 1; ++p) {
    int t0 = 2 * p, t1 = t0 + 1;
    int kh0 = t0 / KSZ, kw0 = t0 % KSZ;
    int kh1 = t1 / KSZ, kw1 = t1 % KSZ;
    // shared LDS index math for both patches
    int i0 = ((wv + (half ? kh1 : kh0)) * PCW + (m + (half ? kw1 : kw0))) * NC;
    int i1 = i0 + 8;
    union { v16bf v; v8bf h[2]; } aM, aW;
    aM.h[0] = *reinterpret_cast<const v8bf*>(&patch[i0]);
    aM.h[1] = *reinterpret_cast<const v8bf*>(&patch[i1]);
    aW.h[0] = *reinterpret_cast<const v8bf*>(&patchD[i0]);
    aW.h[1] = *reinterpret_cast<const v8bf*>(&patchD[i1]);
    size_t fo = (size_t)p * 512 + lane * 16;
    v16bf bM = *reinterpret_cast<const v16bf*>(fragM + fo);
    v16bf bW = *reinterpret_cast<const v16bf*>(fragW + fo);
    __builtin_prefetch(fragM + fo + 512, 0, 0);
    __builtin_prefetch(fragW + fo + 512, 0, 0);
    accM = __builtin_amdgcn_wmma_f32_16x16x32_bf16(false, aM.v, false, bM, (short)0,
                                                   accM, false, false);
    accW = __builtin_amdgcn_wmma_f32_16x16x32_bf16(false, aW.v, false, bW, (short)0,
                                                   accW, false, false);
  }
  {   // epilogue pair: tap 440 valid, tap 441 zero (B fragments zeroed by k_frags)
    int p = NPAIRS - 1;
    int t0 = 2 * p;
    int kh0 = t0 / KSZ, kw0 = t0 % KSZ;
    int i0 = ((wv + kh0) * PCW + (m + kw0)) * NC;
    union { v8bf v; unsigned u[4]; } z;
    z.u[0] = z.u[1] = z.u[2] = z.u[3] = 0u;
    union { v16bf v; v8bf h[2]; } aM, aW;
    aM.h[0] = *reinterpret_cast<const v8bf*>(&patch[i0]);
    aM.h[1] = *reinterpret_cast<const v8bf*>(&patch[i0 + 8]);
    aW.h[0] = *reinterpret_cast<const v8bf*>(&patchD[i0]);
    aW.h[1] = *reinterpret_cast<const v8bf*>(&patchD[i0 + 8]);
    size_t fo = (size_t)p * 512 + lane * 16;
    v16bf bM = *reinterpret_cast<const v16bf*>(fragM + fo);
    v16bf bW = *reinterpret_cast<const v16bf*>(fragW + fo);
    accM = __builtin_amdgcn_wmma_f32_16x16x32_bf16(false, aM.v, false, bM, (short)0,
                                                   accM, false, false);
    accW = __builtin_amdgcn_wmma_f32_16x16x32_bf16(false, aW.v, false, bW, (short)0,
                                                   accW, false, false);
  }

  // D layout: lane n = lane%16 (cout); element r -> pixel m = r + 8*half.
  // Single base-address computation; per-r offsets are immediates (r * 16 floats).
  union { v8f v; float e[8]; } cm, cw;
  cm.v = accM;
  cw.v = accW;
  int y = y0 + wv;
  float* po = out + (((size_t)b * NH + y) * NW + (x0 + 8 * half)) * NC + (lane & 15);
#pragma unroll
  for (int r = 0; r < 8; ++r) {
    po[r * NC] = cm.e[r] - cw.e[r];   // out = convM - 1.0 * convW
  }
}

// ---------------- Host launcher ----------------
extern "C" void kernel_launch(void* const* d_in, const int* in_sizes, int n_in,
                              void* d_out, int out_size, void* d_ws, size_t ws_size,
                              hipStream_t stream) {
  const float* x  = (const float*)d_in[0];   // (8,256,256,16) f32
  const float* gm = (const float*)d_in[1];   // (16,16) f32
  const float* gw = (const float*)d_in[2];   // (16,16) f32
  float* out = (float*)d_out;

  char* ws = (char*)d_ws;
  size_t off = 0;
  auto wsalloc = [&](size_t bytes) -> void* {
    void* p = ws + off;
    off += (bytes + 255) & ~size_t(255);
    return p;
  };
  __bf16* xbf   = (__bf16*)wsalloc((size_t)NB * NH * NW * NC * 2);   // 16.8 MB
  __bf16* dbf   = (__bf16*)wsalloc((size_t)NB * NH * NW * NC * 2);   // 16.8 MB
  __bf16* frags = (__bf16*)wsalloc((size_t)2 * NPAIRS * 512 * 2);    // 452 KB
  float*  s     = (float*)wsalloc((size_t)NB * NH * NW * 4);         // 2 MB
  float*  R     = (float*)wsalloc((size_t)NB * NH * 4);
  float*  Cc    = (float*)wsalloc((size_t)NB * NW * 4);
  float*  T     = (float*)wsalloc((size_t)NB * 4);
  float*  nI    = (float*)wsalloc(2 * 256 * 4);

  k_norms<<<dim3(2), 256, 0, stream>>>(gm, gw, nI);
  k_frags<<<dim3((2 * NPAIRS * 512 + 255) / 256), 256, 0, stream>>>(gm, gw, nI, frags);
  k_stats1<<<dim3(NH, NB), 256, 0, stream>>>(x, xbf, s, R);
  k_stats2<<<dim3(NB), 256, 0, stream>>>(s, Cc, T);
  k_diff<<<dim3(NH, NB), 256, 0, stream>>>(x, s, R, Cc, T, dbf);
  // out = conv(x, Km) - conv(diff, Kw), fused single pass
  k_conv_fused<<<dim3(NW / 16, NH / 8, NB), 256, 0, stream>>>(
      xbf, dbf, frags, frags + (size_t)NPAIRS * 512, out);
}
